// BlockwiseParallelMoE_35390530519957
// MI455X (gfx1250) — matrix operand order repaired and checked by reading, not compiled, not run on hardware
//
#include <hip/hip_runtime.h>
#include <hip/hip_bf16.h>
#include <math.h>

typedef __attribute__((ext_vector_type(16))) __bf16 v16bf;
typedef __attribute__((ext_vector_type(4)))  __bf16 v4bf;
typedef __attribute__((ext_vector_type(8)))  float  v8f;
typedef __attribute__((ext_vector_type(4)))  unsigned int u32x4;
typedef __attribute__((ext_vector_type(8)))  int    i32x8;
typedef __attribute__((ext_vector_type(4)))  int    i32x4;

#define NTOK   8192
#define DMODEL 1024
#define DFF    4096
#define NEXP   8
#define KCAP   2048
#define LDSPAD 40   // padded K-stride (elements): 32 data + 8 pad (64B + 16B per row)

#if defined(__has_builtin)
#if __has_builtin(__builtin_amdgcn_tensor_load_to_lds) && __has_builtin(__builtin_amdgcn_s_wait_tensorcnt)
#define USE_TDM 1
#else
#define USE_TDM 0
#endif
#else
#define USE_TDM 0
#endif

// ---------------------------------------------------------------- zero output
__global__ void moe_zero_kernel(float* __restrict__ out, int n4) {
    int i = blockIdx.x * blockDim.x + threadIdx.x;
    if (i < n4) ((float4*)out)[i] = make_float4(0.f, 0.f, 0.f, 0.f);
}

// ---------------------------------------------------------------- router
__global__ __launch_bounds__(256) void moe_router_kernel(
    const float* __restrict__ x, const float* __restrict__ Wr,
    float* __restrict__ scoresT) {
    int lane  = threadIdx.x & 31;
    int wave  = threadIdx.x >> 5;
    int token = blockIdx.x * 8 + wave;
    const float* xr = x + (size_t)token * DMODEL;

    float acc[NEXP];
#pragma unroll
    for (int e = 0; e < NEXP; ++e) acc[e] = 0.f;

    for (int i = 0; i < DMODEL / 32; ++i) {
        int d = i * 32 + lane;
        float xv = xr[d];
        const float* wrow = Wr + (size_t)d * NEXP;
#pragma unroll
        for (int e = 0; e < NEXP; ++e) acc[e] += xv * wrow[e];
    }
#pragma unroll
    for (int off = 16; off > 0; off >>= 1)
#pragma unroll
        for (int e = 0; e < NEXP; ++e) acc[e] += __shfl_xor(acc[e], off, 32);

    float m = acc[0];
#pragma unroll
    for (int e = 1; e < NEXP; ++e) m = fmaxf(m, acc[e]);
    float s = 0.f;
#pragma unroll
    for (int e = 0; e < NEXP; ++e) { acc[e] = __expf(acc[e] - m); s += acc[e]; }
    float inv = 1.f / s;
    if (lane < NEXP) scoresT[(size_t)lane * NTOK + token] = acc[lane] * inv;
}

// ---------------------------------------------------------------- per-expert top-k
__global__ __launch_bounds__(1024) void moe_topk_kernel(
    const float* __restrict__ scoresT, float* __restrict__ G, int* __restrict__ IDX) {
    __shared__ float key[NTOK];
    __shared__ int   val[NTOK];
    int e = blockIdx.x, tid = threadIdx.x;
    for (int i = tid; i < NTOK; i += 1024) { key[i] = scoresT[(size_t)e * NTOK + i]; val[i] = i; }
    __syncthreads();
    for (int k = 2; k <= NTOK; k <<= 1) {
        for (int j = k >> 1; j > 0; j >>= 1) {
            for (int t = tid; t < NTOK; t += 1024) {
                int ixj = t ^ j;
                if (ixj > t) {
                    bool descending = ((t & k) == 0);
                    float a = key[t], b = key[ixj];
                    if ((a < b) == descending) {
                        key[t] = b; key[ixj] = a;
                        int tv = val[t]; val[t] = val[ixj]; val[ixj] = tv;
                    }
                }
            }
            __syncthreads();
        }
    }
    for (int i = tid; i < KCAP; i += 1024) {
        G[(size_t)e * KCAP + i]   = key[i];
        IDX[(size_t)e * KCAP + i] = val[i];
    }
}

// ---------------------------------------------------------------- gather + fp32->bf16
__global__ __launch_bounds__(256) void moe_gather_kernel(
    const float* __restrict__ x, const int* __restrict__ IDX, __bf16* __restrict__ Xin) {
    int row = blockIdx.x;               // e*KCAP + slot
    int token = IDX[row];
    const float4* src = (const float4*)(x + (size_t)token * DMODEL);
    __bf16* dst = Xin + (size_t)row * DMODEL;
    int t = threadIdx.x;
    float4 v = src[t];
    v4bf o; o[0] = (__bf16)v.x; o[1] = (__bf16)v.y; o[2] = (__bf16)v.z; o[3] = (__bf16)v.w;
    *(v4bf*)(dst + t * 4) = o;
}

// ---------------------------------------------------------------- TDM tile load (bf16 2D tile -> padded LDS)
#if USE_TDM
__device__ inline void tdm_load_tile_bf16(const __bf16* gsrc, unsigned lds_byte_addr,
                                          unsigned row_stride_elems) {
    unsigned long long ga = (unsigned long long)(uintptr_t)gsrc;
    u32x4 g0;
    g0[0] = 1u;                                                  // count=1, user mode
    g0[1] = lds_byte_addr;                                       // LDS dest (bytes)
    g0[2] = (unsigned)ga;                                        // global addr lo
    g0[3] = ((unsigned)(ga >> 32) & 0x1FFFFFFu) | 0x80000000u;   // addr hi | type=2
    i32x8 g1;
    // data_size=1 (2B) | pad_enable | pad_interval=3 (16 DW = 64B row) | pad_amount=3 (4 DW = 16B)
    g1[0] = (1 << 16) | (1 << 20) | (3 << 22) | (3 << 25);
    g1[1] = (int)(32u << 16);          // tensor_dim0 = 32 (bits 79:48 lo half)
    g1[2] = (int)(128u << 16);         // tensor_dim0 hi = 0, tensor_dim1 = 128 (lo half)
    g1[3] = (int)(32u << 16);          // tensor_dim1 hi = 0, tile_dim0 = 32
    g1[4] = 128;                       // tile_dim1 = 128, tile_dim2 = 0
    g1[5] = (int)row_stride_elems;     // tensor_dim0_stride (data_size units)
    g1[6] = 0;
    g1[7] = 0;
    i32x4 gz4 = {0, 0, 0, 0};
    i32x8 gz8 = {0, 0, 0, 0, 0, 0, 0, 0};
    // amdgpu-toolchain (clang-23 / therock-10.0) 6-arg form
    __builtin_amdgcn_tensor_load_to_lds(g0, g1, gz4, gz4, gz8, 0);
}
#endif

// ---------------------------------------------------------------- staging helpers
struct Aregs { uint4 a0, a1; };
__device__ inline Aregs load_a_tile(const __bf16* A, int lda, int m0, int kb, int tid) {
    int row = tid >> 1, c = (tid & 1) * 16;
    const uint4* g = (const uint4*)(A + (size_t)(m0 + row) * lda + kb + c);
    Aregs r; r.a0 = g[0]; r.a1 = g[1];
    return r;
}
__device__ inline void store_a_tile(__bf16* Abuf, const Aregs& r, int tid) {
    int row = tid >> 1, c = (tid & 1) * 16;
    *(uint4*)(&Abuf[row * LDSPAD + c])     = r.a0;
    *(uint4*)(&Abuf[row * LDSPAD + c + 8]) = r.a1;
}

struct Bregs { float4 r[4]; };
__device__ inline Bregs load_b_tile(const float* W, int ldb, int n0, int kb, int tid) {
    int kg = (tid >> 5) * 4;       // k-group base: 0,4,...,28
    int c  = (tid & 31) * 4;       // 4 consecutive cols
    Bregs o;
#pragma unroll
    for (int r = 0; r < 4; ++r)
        o.r[r] = *(const float4*)(W + (size_t)(kb + kg + r) * ldb + n0 + c);
    return o;
}
__device__ inline void store_b_tile(__bf16* Bbuf, const Bregs& o, int tid) {
    int kg = (tid >> 5) * 4;
    int c  = (tid & 31) * 4;
#pragma unroll
    for (int j = 0; j < 4; ++j) {   // col j: pack 4 k-values -> one b64 store
        v4bf p;
        p[0] = (__bf16)((&o.r[0].x)[j]);
        p[1] = (__bf16)((&o.r[1].x)[j]);
        p[2] = (__bf16)((&o.r[2].x)[j]);
        p[3] = (__bf16)((&o.r[3].x)[j]);
        *(v4bf*)(&Bbuf[(c + j) * LDSPAD + kg]) = p;
    }
}

// ---------------------------------------------------------------- WMMA fragment helpers
__device__ inline v16bf frag_a(const __bf16* As, int row, int hi) {
    const __bf16* p = As + row * LDSPAD + hi * 16;
    v16bf a;
#pragma unroll
    for (int i = 0; i < 16; ++i) a[i] = p[i];
    return a;
}
__device__ inline v16bf frag_b(const __bf16* Bs, int col, int hi) {
    const __bf16* p = Bs + col * LDSPAD + hi * 16;
    v16bf b;
#pragma unroll
    for (int i = 0; i < 16; ++i) b[i] = p[i];
    return b;
}

// ---------------------------------------------------------------- GEMM1: H = gelu(Xin @ W1 + b1)
__global__ __launch_bounds__(256) void moe_gemm1_kernel(
    const __bf16* __restrict__ XinE, const float* __restrict__ W1e,
    const float* __restrict__ b1e, __bf16* __restrict__ H) {
    __shared__ __bf16 As[2][128 * LDSPAD];
    __shared__ __bf16 Bs[2][128 * LDSPAD];

    int n0 = blockIdx.x * 128, m0 = blockIdx.y * 128;
    int tid = threadIdx.x, lane = tid & 31, wave = tid >> 5;
    int wm = wave >> 2, wn = wave & 3;       // 2x4 wave grid, each wave 64x32
    int lr = lane & 15, hi = lane >> 4;

    v8f acc[4][2] = {};

    // prologue: stage tile kb=0 into buffer 0
#if USE_TDM
    if (tid < 32) tdm_load_tile_bf16(XinE + (size_t)m0 * DMODEL,
                                     (unsigned)(uintptr_t)&As[0][0], DMODEL);
#else
    { Aregs ar = load_a_tile(XinE, DMODEL, m0, 0, tid); store_a_tile(As[0], ar, tid); }
#endif
    { Bregs br = load_b_tile(W1e, DFF, n0, 0, tid); store_b_tile(Bs[0], br, tid); }
#if USE_TDM
    if (tid < 32) __builtin_amdgcn_s_wait_tensorcnt(0);
#endif
    __syncthreads();

    for (int kb = 0; kb < DMODEL; kb += 32) {
        int p = (kb >> 5) & 1;
        bool more = (kb + 32) < DMODEL;

        // issue next tile's global traffic before computing on the current one
#if USE_TDM
        if (more && tid < 32)
            tdm_load_tile_bf16(XinE + (size_t)m0 * DMODEL + kb + 32,
                               (unsigned)(uintptr_t)&As[p ^ 1][0], DMODEL);
        Aregs ar;
#else
        Aregs ar = {};
        if (more) ar = load_a_tile(XinE, DMODEL, m0, kb + 32, tid);
#endif
        Bregs br = {};
        if (more) br = load_b_tile(W1e, DFF, n0, kb + 32, tid);
        if (more) __builtin_prefetch(W1e + (size_t)(kb + 64) * DFF + n0, 0, 1);

        // compute current buffer
        v16bf af[4], bf[2];
#pragma unroll
        for (int mf = 0; mf < 4; ++mf) af[mf] = frag_a(As[p], wm * 64 + mf * 16 + lr, hi);
#pragma unroll
        for (int nf = 0; nf < 2; ++nf) bf[nf] = frag_b(Bs[p], wn * 32 + nf * 16 + lr, hi);
#pragma unroll
        for (int mf = 0; mf < 4; ++mf)
#pragma unroll
            for (int nf = 0; nf < 2; ++nf)
                acc[mf][nf] = __builtin_amdgcn_wmma_f32_16x16x32_bf16(
                    false, af[mf], false, bf[nf], (short)0, acc[mf][nf], false, false);

        // drain staged registers into the other buffer
#if !USE_TDM
        if (more) store_a_tile(As[p ^ 1], ar, tid);
#else
        (void)ar;
#endif
        if (more) store_b_tile(Bs[p ^ 1], br, tid);
#if USE_TDM
        if (more && tid < 32) __builtin_amdgcn_s_wait_tensorcnt(0);
#endif
        __syncthreads();
    }

    // epilogue: bias + exact GELU -> bf16 H
#pragma unroll
    for (int nf = 0; nf < 2; ++nf) {
        int col = n0 + wn * 32 + nf * 16 + lr;
        float bias = b1e[col];
#pragma unroll
        for (int mf = 0; mf < 4; ++mf) {
#pragma unroll
            for (int r = 0; r < 8; ++r) {
                int row = m0 + wm * 64 + mf * 16 + hi * 8 + r;
                float v = acc[mf][nf][r] + bias;
                v = 0.5f * v * (1.0f + erff(v * 0.70710678118f));
                H[(size_t)row * DFF + col] = (__bf16)v;
            }
        }
    }
}

// ---------------------------------------------------------------- GEMM2: out[idx[m]] += G[m]*(H @ W2 + b2)
__global__ __launch_bounds__(256) void moe_gemm2_kernel(
    const __bf16* __restrict__ H, const float* __restrict__ W2e,
    const float* __restrict__ b2e, const int* __restrict__ IDXe,
    const float* __restrict__ Ge, float* __restrict__ out) {
    __shared__ __bf16 As[2][128 * LDSPAD];
    __shared__ __bf16 Bs[2][128 * LDSPAD];

    int n0 = blockIdx.x * 128, m0 = blockIdx.y * 128;
    int tid = threadIdx.x, lane = tid & 31, wave = tid >> 5;
    int wm = wave >> 2, wn = wave & 3;
    int lr = lane & 15, hi = lane >> 4;

    v8f acc[4][2] = {};

#if USE_TDM
    if (tid < 32) tdm_load_tile_bf16(H + (size_t)m0 * DFF,
                                     (unsigned)(uintptr_t)&As[0][0], DFF);
#else
    { Aregs ar = load_a_tile(H, DFF, m0, 0, tid); store_a_tile(As[0], ar, tid); }
#endif
    { Bregs br = load_b_tile(W2e, DMODEL, n0, 0, tid); store_b_tile(Bs[0], br, tid); }
#if USE_TDM
    if (tid < 32) __builtin_amdgcn_s_wait_tensorcnt(0);
#endif
    __syncthreads();

    for (int kb = 0; kb < DFF; kb += 32) {
        int p = (kb >> 5) & 1;
        bool more = (kb + 32) < DFF;

#if USE_TDM
        if (more && tid < 32)
            tdm_load_tile_bf16(H + (size_t)m0 * DFF + kb + 32,
                               (unsigned)(uintptr_t)&As[p ^ 1][0], DFF);
        Aregs ar;
#else
        Aregs ar = {};
        if (more) ar = load_a_tile(H, DFF, m0, kb + 32, tid);
#endif
        Bregs br = {};
        if (more) br = load_b_tile(W2e, DMODEL, n0, kb + 32, tid);
        if (more) __builtin_prefetch(W2e + (size_t)(kb + 64) * DMODEL + n0, 0, 1);

        v16bf af[4], bf[2];
#pragma unroll
        for (int mf = 0; mf < 4; ++mf) af[mf] = frag_a(As[p], wm * 64 + mf * 16 + lr, hi);
#pragma unroll
        for (int nf = 0; nf < 2; ++nf) bf[nf] = frag_b(Bs[p], wn * 32 + nf * 16 + lr, hi);
#pragma unroll
        for (int mf = 0; mf < 4; ++mf)
#pragma unroll
            for (int nf = 0; nf < 2; ++nf)
                acc[mf][nf] = __builtin_amdgcn_wmma_f32_16x16x32_bf16(
                    false, af[mf], false, bf[nf], (short)0, acc[mf][nf], false, false);

#if !USE_TDM
        if (more) store_a_tile(As[p ^ 1], ar, tid);
#else
        (void)ar;
#endif
        if (more) store_b_tile(Bs[p ^ 1], br, tid);
#if USE_TDM
        if (more && tid < 32) __builtin_amdgcn_s_wait_tensorcnt(0);
#endif
        __syncthreads();
    }

    // epilogue: bias, gate, scatter-add into output (idx/G loaded once per row)
    float bias0 = b2e[n0 + wn * 32 + lr];
    float bias1 = b2e[n0 + wn * 32 + 16 + lr];
#pragma unroll
    for (int mf = 0; mf < 4; ++mf) {
#pragma unroll
        for (int r = 0; r < 8; ++r) {
            int row = m0 + wm * 64 + mf * 16 + hi * 8 + r;
            int token = IDXe[row];
            float g = Ge[row];
            float* orow = out + (size_t)token * DMODEL;
            float v0 = acc[mf][0][r] + bias0;
            float v1 = acc[mf][1][r] + bias1;
            unsafeAtomicAdd(orow + n0 + wn * 32 + lr,      g * v0);
            unsafeAtomicAdd(orow + n0 + wn * 32 + 16 + lr, g * v1);
        }
    }
}

// ---------------------------------------------------------------- host launcher
extern "C" void kernel_launch(void* const* d_in, const int* in_sizes, int n_in,
                              void* d_out, int out_size, void* d_ws, size_t ws_size,
                              hipStream_t stream) {
    const float* x  = (const float*)d_in[0];
    const float* Wr = (const float*)d_in[1];
    const float* W1 = (const float*)d_in[2];
    const float* b1 = (const float*)d_in[3];
    const float* W2 = (const float*)d_in[4];
    const float* b2 = (const float*)d_in[5];
    float* out = (float*)d_out;

    char* ws = (char*)d_ws;
    float*  scoresT = (float*)(ws);                                   // 256 KB
    int*    IDX     = (int*)(ws + 262144);                            // 64 KB
    float*  G       = (float*)(ws + 327680);                          // 64 KB
    __bf16* Xin     = (__bf16*)(ws + 393216);                         // 32 MB
    __bf16* H       = (__bf16*)(ws + 393216 + (size_t)NEXP * KCAP * DMODEL * 2); // 16 MB, reused per expert

    int n4 = out_size / 4;
    moe_zero_kernel<<<(n4 + 255) / 256, 256, 0, stream>>>(out, n4);
    moe_router_kernel<<<NTOK / 8, 256, 0, stream>>>(x, Wr, scoresT);
    moe_topk_kernel<<<NEXP, 1024, 0, stream>>>(scoresT, G, IDX);
    moe_gather_kernel<<<NEXP * KCAP, 256, 0, stream>>>(x, IDX, Xin);

    for (int e = 0; e < NEXP; ++e) {
        const __bf16* XinE = Xin + (size_t)e * KCAP * DMODEL;
        const float*  W1e  = W1 + (size_t)e * DMODEL * DFF;
        const float*  b1e  = b1 + (size_t)e * DFF;
        const float*  W2e  = W2 + (size_t)e * DFF * DMODEL;
        const float*  b2e  = b2 + (size_t)e * DMODEL;
        moe_gemm1_kernel<<<dim3(DFF / 128, KCAP / 128), 256, 0, stream>>>(XinE, W1e, b1e, H);
        moe_gemm2_kernel<<<dim3(DMODEL / 128, KCAP / 128), 256, 0, stream>>>(
            H, W2e, b2e, IDX + e * KCAP, G + e * KCAP, out);
    }
}
